// GCN_39883066310783
// MI455X (gfx1250) — matrix operand order repaired and checked by reading, not compiled, not run on hardware
//
#include <hip/hip_runtime.h>
#include <hip/hip_bf16.h>

typedef unsigned short u16;
typedef unsigned int u32;
typedef __attribute__((ext_vector_type(16))) __bf16 bf16x16;
typedef __attribute__((ext_vector_type(8)))  float  f32x8;

union FragU { uint4 u[2]; bf16x16 v; };

__device__ __forceinline__ u16 f32_to_bf16(float f) {
    u32 u = __float_as_uint(f);
    u32 r = u + 0x7FFFu + ((u >> 16) & 1u);   // round-to-nearest-even
    return (u16)(r >> 16);
}
__device__ __forceinline__ float bf16_to_f32(u16 h) {
    return __uint_as_float(((u32)h) << 16);
}
// monotone float<->uint mapping for atomicMax on floats
__device__ __forceinline__ u32 fkey(float f) {
    u32 b = __float_as_uint(f);
    return (b & 0x80000000u) ? ~b : (b | 0x80000000u);
}
__device__ __forceinline__ float funkey(u32 k) {
    return __uint_as_float((k & 0x80000000u) ? (k & 0x7FFFFFFFu) : ~k);
}

// ---------------------------------------------------------------- cvt fp32 -> bf16
__global__ void gat_cvt_bf16(const float* __restrict__ x, u16* __restrict__ y, size_t n) {
    size_t i = (size_t)blockIdx.x * blockDim.x + threadIdx.x;
    if (i < n) y[i] = f32_to_bf16(x[i]);
}

// ---------------------------------------------------------------- W [K x Nout] -> WT bf16 [Nout x K]
__global__ void gat_transpose_cvt(const float* __restrict__ W, u16* __restrict__ WT,
                                  int K, int Nout) {
    __shared__ float tile[32][33];
    int bx = blockIdx.x * 32;   // along Nout
    int by = blockIdx.y * 32;   // along K
    int tx = threadIdx.x, ty = threadIdx.y;   // block (32,8)
    for (int r = ty; r < 32; r += 8) {
        int k = by + r, n = bx + tx;
        tile[r][tx] = (k < K && n < Nout) ? W[(size_t)k * Nout + n] : 0.0f;
    }
    __syncthreads();
    for (int r = ty; r < 32; r += 8) {
        int n = bx + r, k = by + tx;
        if (n < Nout && k < K) WT[(size_t)n * K + k] = f32_to_bf16(tile[tx][r]);
    }
}

// ---------------------------------------------------------------- WMMA GEMM: C = A * WT^T (bf16 in, bf16 out, f32 acc)
#define BM 128
#define BN 128
#define BKK 64
#define LDT 72   // padded LDS row stride (elements)

__global__ __launch_bounds__(256) void gat_gemm_bf16_wmma(
    const u16* __restrict__ A,    // [M x K] bf16 row-major
    const u16* __restrict__ BT,   // [Nout x K] bf16 row-major (W transposed)
    u16* __restrict__ C,          // [M x Nout] bf16
    int M, int K, int Nout)
{
    __shared__ u16 sA[BM * LDT];
    __shared__ u16 sB[BN * LDT];

    const int tid  = threadIdx.x;
    const int lane = tid & 31;
    const int wv   = tid >> 5;
    const int wm   = (wv >> 2) * 64;   // wave M offset (0 / 64)
    const int wn   = (wv & 3)  * 32;   // wave N offset (0/32/64/96)
    const int rowBase = blockIdx.y * BM;
    const int colBase = blockIdx.x * BN;

    f32x8 acc[4][2];
    #pragma unroll
    for (int i = 0; i < 4; ++i)
        #pragma unroll
        for (int j = 0; j < 2; ++j)
            #pragma unroll
            for (int v = 0; v < 8; ++v) acc[i][j][v] = 0.0f;

    const int lr = tid >> 3;          // 0..31
    const int lc = (tid & 7) << 3;    // 0..56 (8 bf16 = 16B per load)

    for (int k0 = 0; k0 < K; k0 += BKK) {
        __syncthreads();
        #pragma unroll
        for (int p = 0; p < 4; ++p) {
            int r  = p * 32 + lr;
            uint4 va = make_uint4(0u, 0u, 0u, 0u);
            int gr = rowBase + r;
            if (gr < M) va = *(const uint4*)(A + (size_t)gr * K + k0 + lc);
            *(uint4*)&sA[r * LDT + lc] = va;
            uint4 vb = make_uint4(0u, 0u, 0u, 0u);
            int gn = colBase + r;
            if (gn < Nout) vb = *(const uint4*)(BT + (size_t)gn * K + k0 + lc);
            *(uint4*)&sB[r * LDT + lc] = vb;
        }
        __syncthreads();

        #pragma unroll
        for (int c = 0; c < 2; ++c) {
            const int kb   = c * 32;
            const int arow = lane & 15;
            // A frag: per-lane K in {abase..abase+7} U {abase+16..abase+23}
            const int abase = kb + ((lane >> 4) << 3);
            bf16x16 af[4];
            #pragma unroll
            for (int i = 0; i < 4; ++i) {
                FragU fu;
                const u16* pa = &sA[(wm + i * 16 + arow) * LDT + abase];
                fu.u[0] = *(const uint4*)pa;
                fu.u[1] = *(const uint4*)(pa + 16);
                af[i] = fu.v;
            }
            // B frag: per-lane K in {bbase..bbase+7} U {bbase+16..bbase+23}
            const int bbase = kb + ((lane >> 4) << 3);
            bf16x16 bfr[2];
            #pragma unroll
            for (int j = 0; j < 2; ++j) {
                FragU fu;
                const u16* pb = &sB[(wn + j * 16 + arow) * LDT + bbase];
                fu.u[0] = *(const uint4*)pb;
                fu.u[1] = *(const uint4*)(pb + 16);
                bfr[j] = fu.v;
            }
            #pragma unroll
            for (int i = 0; i < 4; ++i)
                #pragma unroll
                for (int j = 0; j < 2; ++j)
                    acc[i][j] = __builtin_amdgcn_wmma_f32_16x16x32_bf16(
                        false, af[i], false, bfr[j], (short)0, acc[i][j], false, false);
        }
    }

    // epilogue: C element (M = v + 8*(lane>>4), N = lane&15)
    const int mhalf = (lane >> 4) << 3;
    const int nloc  = lane & 15;
    #pragma unroll
    for (int i = 0; i < 4; ++i)
        #pragma unroll
        for (int j = 0; j < 2; ++j)
            #pragma unroll
            for (int v = 0; v < 8; ++v) {
                int gr = rowBase + wm + i * 16 + mhalf + v;
                int gc = colBase + wn + j * 16 + nloc;
                if (gr < M && gc < Nout)
                    C[(size_t)gr * Nout + gc] = f32_to_bf16(acc[i][j][v]);
            }
}

// ---------------------------------------------------------------- attention logits el/er
__global__ void gat_attn_logits(const u16* __restrict__ feat,
                                const float* __restrict__ al, const float* __restrict__ ar,
                                float* __restrict__ el, float* __restrict__ er,
                                int H, int D) {
    int n    = blockIdx.x;
    int wv   = threadIdx.x >> 5;
    int lane = threadIdx.x & 31;
    for (int h = wv; h < H; h += (blockDim.x >> 5)) {
        float sl = 0.0f, sr = 0.0f;
        const u16*   f   = feat + (size_t)n * H * D + (size_t)h * D;
        const float* alh = al + (size_t)h * D;
        const float* arh = ar + (size_t)h * D;
        for (int d = lane; d < D; d += 32) {
            float v = bf16_to_f32(f[d]);
            sl += v * alh[d];
            sr += v * arh[d];
        }
        #pragma unroll
        for (int off = 16; off > 0; off >>= 1) {
            sl += __shfl_xor(sl, off, 32);
            sr += __shfl_xor(sr, off, 32);
        }
        if (lane == 0) { el[n * H + h] = sl; er[n * H + h] = sr; }
    }
}

// ---------------------------------------------------------------- init segment-max keys / sums
__global__ void gat_init_ms(u32* __restrict__ mkey, float* __restrict__ s, int n) {
    int i = blockIdx.x * blockDim.x + threadIdx.x;
    if (i < n) { mkey[i] = 0x007FFFFFu; /* key(-inf) */ s[i] = 0.0f; }
}

// ---------------------------------------------------------------- edge pass 1: leaky_relu + segment max
__global__ void gat_edge_max(const int* __restrict__ src, const int* __restrict__ dst,
                             const float* __restrict__ el, const float* __restrict__ er,
                             float* __restrict__ ebuf, u32* __restrict__ mkey,
                             int EH, int H) {
    int i = blockIdx.x * blockDim.x + threadIdx.x;
    if (i >= EH) return;
    int e = i / H, h = i - e * H;
    float v = el[src[e] * H + h] + er[dst[e] * H + h];
    v = (v > 0.0f) ? v : 0.2f * v;          // leaky relu, DGL slope
    ebuf[i] = v;
    atomicMax(&mkey[dst[e] * H + h], fkey(v));
}

// ---------------------------------------------------------------- edge pass 2: exp + segment sum
__global__ void gat_edge_exp(const int* __restrict__ dst,
                             float* __restrict__ ebuf, const u32* __restrict__ mkey,
                             float* __restrict__ s, int EH, int H) {
    int i = blockIdx.x * blockDim.x + threadIdx.x;
    if (i >= EH) return;
    int e = i / H, h = i - e * H;
    float m = funkey(mkey[dst[e] * H + h]);
    if (!(m > -3.0e38f)) m = 0.0f;          // -inf (no-edge) -> 0 per reference
    float ee = __expf(ebuf[i] - m);
    ebuf[i] = ee;
    atomicAdd(&s[dst[e] * H + h], ee);
}

// ---------------------------------------------------------------- edge pass 3: scatter-add a*feat[src] -> out[dst]
__global__ __launch_bounds__(256) void gat_edge_aggregate(
    const int* __restrict__ src, const int* __restrict__ dst,
    const float* __restrict__ ebuf, const float* __restrict__ s,
    const u16* __restrict__ feat, float* __restrict__ out,
    int H, int logD) {
    __shared__ float coef[8];
    int e   = blockIdx.x;
    int tid = threadIdx.x;
    int sn = src[e], dn = dst[e];
    if (tid < H) {
        float sv = s[dn * H + tid];
        coef[tid] = ebuf[e * H + tid] / ((sv == 0.0f) ? 1.0f : sv);
    }
    __syncthreads();
    int HD = H << logD;
    const u16* fs = feat + (size_t)sn * HD;
    float*     od = out  + (size_t)dn * HD;
    for (int i = tid; i < HD; i += blockDim.x)
        atomicAdd(&od[i], coef[i >> logD] * bf16_to_f32(fs[i]));
}

// ---------------------------------------------------------------- bias + relu + cvt -> next layer input (bf16)
__global__ void gat_bias_relu_cvt(const float* __restrict__ agg, const float* __restrict__ b,
                                  u16* __restrict__ nxt, int hdMask, size_t total) {
    size_t i = (size_t)blockIdx.x * blockDim.x + threadIdx.x;
    if (i >= total) return;
    float v = agg[i] + b[(int)(i & (size_t)hdMask)];
    v = (v > 0.0f) ? v : 0.0f;
    nxt[i] = f32_to_bf16(v);
}

// ---------------------------------------------------------------- final: column mean + bias -> d_out[64]
__global__ void gat_colmean(const float* __restrict__ agg, const float* __restrict__ b,
                            float* __restrict__ out, int rows, int cols) {
    __shared__ float red[256];
    int c = blockIdx.x;
    float sum = 0.0f;
    for (int r = threadIdx.x; r < rows; r += blockDim.x)
        sum += agg[(size_t)r * cols + c];
    red[threadIdx.x] = sum;
    __syncthreads();
    for (int st = 128; st > 0; st >>= 1) {
        if ((int)threadIdx.x < st) red[threadIdx.x] += red[threadIdx.x + st];
        __syncthreads();
    }
    if (threadIdx.x == 0) out[c] = red[0] / (float)rows + b[c];
}

// ================================================================ host
extern "C" void kernel_launch(void* const* d_in, const int* in_sizes, int n_in,
                              void* d_out, int out_size, void* d_ws, size_t ws_size,
                              hipStream_t stream) {
    const int IN = 128;
    const int Nn = in_sizes[0] / IN;     // 20000
    const int E  = in_sizes[1];          // 100000

    const float* x   = (const float*)d_in[0];
    const int*   src = (const int*)d_in[1];
    const int*   dst = (const int*)d_in[2];
    const float* W[4]  = {(const float*)d_in[3],  (const float*)d_in[7],
                          (const float*)d_in[11], (const float*)d_in[15]};
    const float* al[4] = {(const float*)d_in[4],  (const float*)d_in[8],
                          (const float*)d_in[12], (const float*)d_in[16]};
    const float* ar[4] = {(const float*)d_in[5],  (const float*)d_in[9],
                          (const float*)d_in[13], (const float*)d_in[17]};
    const float* bi[4] = {(const float*)d_in[6],  (const float*)d_in[10],
                          (const float*)d_in[14], (const float*)d_in[18]};

    const int Hh[4]   = {8, 8, 8, 1};
    const int Dd[4]   = {512, 512, 512, 64};
    const int logD[4] = {9, 9, 9, 6};
    const int Kin[4]  = {128, 4096, 4096, 4096};
    const int Kout[4] = {4096, 4096, 4096, 64};

    char* ws = (char*)d_ws;
    size_t off = 0;
    auto carve = [&](size_t bytes) -> void* {
        void* p = (void*)(ws + off);
        off += (bytes + 255) & ~(size_t)255;
        return p;
    };
    u16*   inb   = (u16*)  carve((size_t)Nn * 4096 * sizeof(u16));   // GEMM input (bf16)
    u16*   featb = (u16*)  carve((size_t)Nn * 4096 * sizeof(u16));   // projected features (bf16)
    u16*   WT    = (u16*)  carve((size_t)4096 * 4096 * sizeof(u16)); // transposed weights (bf16)
    float* agg   = (float*)carve((size_t)Nn * 4096 * sizeof(float)); // aggregation accumulator
    float* el    = (float*)carve((size_t)Nn * 8 * sizeof(float));
    float* er    = (float*)carve((size_t)Nn * 8 * sizeof(float));
    u32*   mkey  = (u32*)  carve((size_t)Nn * 8 * sizeof(u32));
    float* s     = (float*)carve((size_t)Nn * 8 * sizeof(float));
    float* ebuf  = (float*)carve((size_t)E * 8 * sizeof(float));
    (void)ws_size; (void)n_in; (void)out_size;

    // layer-0 input: x -> bf16
    {
        size_t tot = (size_t)Nn * IN;
        gat_cvt_bf16<<<(tot + 255) / 256, 256, 0, stream>>>(x, inb, tot);
    }

    for (int L = 0; L < 4; ++L) {
        const int H = Hh[L], D = Dd[L], K = Kin[L], Nout = Kout[L];

        // W -> WT (bf16, transposed)
        {
            dim3 grid((Nout + 31) / 32, (K + 31) / 32), blk(32, 8);
            gat_transpose_cvt<<<grid, blk, 0, stream>>>(W[L], WT, K, Nout);
        }
        // feat = inb @ W  (bf16 WMMA, f32 accumulate)
        {
            dim3 grid((Nout + BN - 1) / BN, (Nn + BM - 1) / BM);
            gat_gemm_bf16_wmma<<<grid, 256, 0, stream>>>(inb, WT, featb, Nn, K, Nout);
        }
        // attention logits
        gat_attn_logits<<<Nn, 256, 0, stream>>>(featb, al[L], ar[L], el, er, H, D);
        // init softmax state + zero aggregation buffer
        {
            int nh = Nn * H;
            gat_init_ms<<<(nh + 255) / 256, 256, 0, stream>>>(mkey, s, nh);
            hipMemsetAsync(agg, 0, (size_t)Nn * Nout * sizeof(float), stream);
        }
        // edge softmax + aggregation
        {
            int EH = E * H;
            gat_edge_max<<<(EH + 255) / 256, 256, 0, stream>>>(src, dst, el, er, ebuf, mkey, EH, H);
            gat_edge_exp<<<(EH + 255) / 256, 256, 0, stream>>>(dst, ebuf, mkey, s, EH, H);
            gat_edge_aggregate<<<E, 256, 0, stream>>>(src, dst, ebuf, s, featb, agg, H, logD[L]);
        }
        if (L < 3) {
            size_t tot = (size_t)Nn * Nout;
            gat_bias_relu_cvt<<<(tot + 255) / 256, 256, 0, stream>>>(agg, bi[L], inb,
                                                                     Nout - 1, tot);
        } else {
            gat_colmean<<<Nout, 256, 0, stream>>>(agg, bi[L], (float*)d_out, Nn, Nout);
        }
    }
}
